// GIN_48696339202587
// MI455X (gfx1250) — matrix operand order repaired and checked by reading, not compiled, not run on hardware
//
#include <hip/hip_runtime.h>
#include <stdint.h>

// =============================================================================
// GIN 2-layer forward for MI455X (gfx1250, wave32, WMMA).
// Precision strategy: bf16 WMMA (v_wmma_f32_16x16x32_bf16) with f32 accumulate
// for all GEMMs; f32 edge aggregation with native fadd atomics; bf16 storage
// for inter-layer activations to halve the dominant scatter/gather traffic.
// =============================================================================

typedef __attribute__((ext_vector_type(16))) __bf16 v16bf;
typedef __attribute__((ext_vector_type(8)))  float  v8f;

#define IN_DIM 128
#define HIDDIM 512
#define NCLS   64
#define BN_EPS 1e-5f

__device__ __forceinline__ uint16_t f2bf(float f) {
  uint32_t u = __float_as_uint(f);
  u += 0x7FFFu + ((u >> 16) & 1u);          // round-to-nearest-even
  return (uint16_t)(u >> 16);
}
__device__ __forceinline__ float bf2f(uint16_t h) {
  return __uint_as_float(((uint32_t)h) << 16);
}
__device__ __forceinline__ uint32_t pack2bf(float a, float b) {
  return (uint32_t)f2bf(a) | ((uint32_t)f2bf(b) << 16);
}
__device__ __forceinline__ void atom_fadd(float* p, float v) {
  __hip_atomic_fetch_add(p, v, __ATOMIC_RELAXED, __HIP_MEMORY_SCOPE_AGENT);
}

union FragBF { uint32_t u[8]; v16bf v; };

// ---------------------------------------------------------------------------
// Weight packing: W[Kin x Kout] f32 -> WMMA B-fragment-ordered bf16 dwords.
// Layout: Wp[((nt*kchunks + kc)*32 + lane)*8 + v], where the dword packs
// W[(kc*32 + (v>>2)*16 + hi*8 + 2(v&3)), nt*16 + (lane&15)] (lo) and K+1 (hi), hi=lane>>4.
// ---------------------------------------------------------------------------
__global__ void pack_w_kernel(const float* __restrict__ W, uint32_t* __restrict__ Wp,
                              int Kin, int Kout) {
  int tid = blockIdx.x * blockDim.x + threadIdx.x;
  int total = (Kin * Kout) >> 1;
  if (tid >= total) return;
  int v      = tid & 7;
  int lane   = (tid >> 3) & 31;
  int chunk  = tid >> 8;                     // nt*kchunks + kc
  int kchunk = Kin >> 5;
  int kc     = chunk % kchunk;
  int nt     = chunk / kchunk;
  int n      = nt * 16 + (lane & 15);
  int hi     = lane >> 4;
  int k0     = kc * 32 + (v >> 2) * 16 + hi * 8 + 2 * (v & 3);
  Wp[tid] = pack2bf(W[(size_t)k0 * Kout + n], W[(size_t)(k0 + 1) * Kout + n]);
}

// ---------------------------------------------------------------------------
// WMMA GEMM: Out[nrows x KOUT] = act(A[nrows x KIN] @ W + bias)
// ABF: A stored as bf16 (else f32, converted on load)
// ACT: 0 = relu, 1 = selu ; OBF: output bf16 (else f32)
// One wave per 16x16 output tile; K swept in chunks of 32.
// ---------------------------------------------------------------------------
template<int KIN, int KOUT, int ABF, int ACT, int OBF>
__global__ __launch_bounds__(256) void gemm_bias_act(
    const void* __restrict__ Aptr, const uint32_t* __restrict__ Wp,
    const float* __restrict__ bias, void* __restrict__ Out, int nrows) {
  const int lane   = threadIdx.x & 31;
  const int wid    = threadIdx.x >> 5;
  const int tile_n = blockIdx.x;
  const int tile_m = blockIdx.y * 8 + wid;
  const int m0     = tile_m * 16;
  if (m0 >= nrows) return;                 // wave-uniform: EXEC stays all-1s

  const int ml = m0 + (lane & 15);
  const int hi = lane >> 4;
  const int kchunks = KIN >> 5;

  v8f c;
#pragma unroll
  for (int i = 0; i < 8; ++i) c[i] = 0.0f;

  for (int kc = 0; kc < kchunks; ++kc) {
    FragBF a, b;
    // B fragment: fully coalesced 32B/lane from the packed weights
    const uint4* wqc = ((const uint4*)Wp) +
        (((size_t)(tile_n * kchunks + kc) * 32 + lane) * 2);
    uint4 b0 = wqc[0], b1 = wqc[1];
    b.u[0] = b0.x; b.u[1] = b0.y; b.u[2] = b0.z; b.u[3] = b0.w;
    b.u[4] = b1.x; b.u[5] = b1.y; b.u[6] = b1.z; b.u[7] = b1.w;

    if (ABF) {
      // A 16x32 bf16 layout: dword idx = (v>>2)*8 + hi*4 + (v&3)
      const uint32_t* arow =
          (const uint32_t*)((const uint16_t*)Aptr + (size_t)ml * KIN) + kc * 16;
      const uint4* ap = (const uint4*)arow;
      uint4 a0 = ap[hi];
      uint4 a1 = ap[2 + hi];
      a.u[0] = a0.x; a.u[1] = a0.y; a.u[2] = a0.z; a.u[3] = a0.w;
      a.u[4] = a1.x; a.u[5] = a1.y; a.u[6] = a1.z; a.u[7] = a1.w;
    } else {
      const float* arow = (const float*)Aptr + (size_t)ml * KIN + kc * 32;
      float4 fa = *(const float4*)(arow + hi * 8);
      float4 fb = *(const float4*)(arow + hi * 8 + 4);
      float4 fc = *(const float4*)(arow + 16 + hi * 8);
      float4 fd = *(const float4*)(arow + 16 + hi * 8 + 4);
      a.u[0] = pack2bf(fa.x, fa.y); a.u[1] = pack2bf(fa.z, fa.w);
      a.u[2] = pack2bf(fb.x, fb.y); a.u[3] = pack2bf(fb.z, fb.w);
      a.u[4] = pack2bf(fc.x, fc.y); a.u[5] = pack2bf(fc.z, fc.w);
      a.u[6] = pack2bf(fd.x, fd.y); a.u[7] = pack2bf(fd.z, fd.w);
    }
    c = __builtin_amdgcn_wmma_f32_16x16x32_bf16(
        false, a.v, false, b.v, (short)0, c, false, false);
  }

  const int ng = tile_n * 16 + (lane & 15);
  const float bv = bias[ng];
#pragma unroll
  for (int r = 0; r < 8; ++r) {
    int mg = m0 + hi * 8 + r;
    float x = c[r] + bv;
    if (ACT == 0) {
      x = fmaxf(x, 0.0f);
    } else {
      const float sc = 1.0507009873554805f, al = 1.6732632423543772f;
      x = (x > 0.0f) ? sc * x : sc * al * (__expf(x) - 1.0f);
    }
    if (OBF) ((uint16_t*)Out)[(size_t)mg * KOUT + ng] = f2bf(x);
    else     ((float*)Out)[(size_t)mg * KOUT + ng] = x;
  }
}

// ---------------------------------------------------------------------------
// Utility kernels
// ---------------------------------------------------------------------------
__global__ void zero_f32(float* p, int n) {
  int tid = blockIdx.x * blockDim.x + threadIdx.x;
  if (tid < n) p[tid] = 0.0f;
}

__global__ void copy_f32x4(const float4* __restrict__ src, float4* __restrict__ dst,
                           long n4) {
  long tid = (long)blockIdx.x * blockDim.x + threadIdx.x;
  if (tid < n4) dst[tid] = src[tid];
}

// layer-1 edge scatter: Acc[dst] += x[src], 4 f32 features per thread
__global__ void scatter_f32(const float* __restrict__ X, const int* __restrict__ src,
                            const int* __restrict__ dst, float* __restrict__ Acc,
                            int E, int F) {
  long tid = (long)blockIdx.x * blockDim.x + threadIdx.x;
  int per = F >> 2;
  long e = tid / per;
  if (e >= E) return;
  int f = (int)(tid % per) * 4;
  int s = src[e], d = dst[e];
  float4 v = *(const float4*)(X + (size_t)s * F + f);
  float* a = Acc + (size_t)d * F + f;
  atom_fadd(a + 0, v.x); atom_fadd(a + 1, v.y);
  atom_fadd(a + 2, v.z); atom_fadd(a + 3, v.w);
}

// layer-2 edge scatter with on-the-fly batchnorm: Acc[dst] += bn(H[src])
__global__ void scatter_bn_bf16(const uint16_t* __restrict__ H,
                                const int* __restrict__ src, const int* __restrict__ dst,
                                const float* __restrict__ scale,
                                const float* __restrict__ shift,
                                float* __restrict__ Acc, int E, int F) {
  long tid = (long)blockIdx.x * blockDim.x + threadIdx.x;
  int per = F >> 2;
  long e = tid / per;
  if (e >= E) return;
  int f = (int)(tid % per) * 4;
  int s = src[e], d = dst[e];
  const uint16_t* h = H + (size_t)s * F + f;
  float* a = Acc + (size_t)d * F + f;
#pragma unroll
  for (int i = 0; i < 4; ++i)
    atom_fadd(a + i, scale[f + i] * bf2f(h[i]) + shift[f + i]);
}

// A2[i] = bn(H1[i]) (self term of GIN layer 2)
__global__ void bn_apply_init(const uint16_t* __restrict__ H,
                              const float* __restrict__ scale,
                              const float* __restrict__ shift,
                              float* __restrict__ A, long total, int F) {
  long idx = ((long)blockIdx.x * blockDim.x + threadIdx.x) * 4;
  if (idx >= total) return;
  int f = (int)(idx % F);
#pragma unroll
  for (int i = 0; i < 4; ++i)
    A[idx + i] = scale[f + i] * bf2f(H[idx + i]) + shift[f + i];
}

// per-feature sum / sumsq via column walkers
__global__ void col_stats_bf16(const uint16_t* __restrict__ H, float* __restrict__ sum,
                               float* __restrict__ sq, int rows, int F, int walkers) {
  int tid = blockIdx.x * blockDim.x + threadIdx.x;
  if (tid >= F * walkers) return;
  int f = tid & (F - 1);
  int w = tid / F;
  float s = 0.0f, q = 0.0f;
  for (int r = w; r < rows; r += walkers) {
    float x = bf2f(H[(size_t)r * F + f]);
    s += x; q += x * x;
  }
  atom_fadd(&sum[f], s);
  atom_fadd(&sq[f], q);
}

__global__ void col_stats_f32(const float* __restrict__ H, float* __restrict__ sum,
                              float* __restrict__ sq, int rows, int F, int walkers) {
  int tid = blockIdx.x * blockDim.x + threadIdx.x;
  if (tid >= F * walkers) return;
  int f = tid & (F - 1);
  int w = tid / F;
  float s = 0.0f, q = 0.0f;
  for (int r = w; r < rows; r += walkers) {
    float x = H[(size_t)r * F + f];
    s += x; q += x * x;
  }
  atom_fadd(&sum[f], s);
  atom_fadd(&sq[f], q);
}

__global__ void bn_finalize(const float* __restrict__ sum, const float* __restrict__ sq,
                            const float* __restrict__ gamma, const float* __restrict__ beta,
                            float* __restrict__ scale, float* __restrict__ shift,
                            int F, float invN) {
  int f = blockIdx.x * blockDim.x + threadIdx.x;
  if (f >= F) return;
  float mu  = sum[f] * invN;
  float var = fmaxf(sq[f] * invN - mu * mu, 0.0f);
  float sc  = gamma[f] * rsqrtf(var + BN_EPS);
  scale[f] = sc;
  shift[f] = beta[f] - mu * sc;
}

// fused BN2 + row softmax over 64 classes: one wave32 per row, 2 elems/lane
__global__ __launch_bounds__(256) void bn_softmax64(
    const float* __restrict__ H, const float* __restrict__ scale,
    const float* __restrict__ shift, float* __restrict__ out, int rows) {
  int lane = threadIdx.x & 31;
  int wid  = threadIdx.x >> 5;
  int row  = blockIdx.x * 8 + wid;
  if (row >= rows) return;
  const float* h = H + (size_t)row * 64;
  float x0 = scale[lane]      * h[lane]      + shift[lane];
  float x1 = scale[lane + 32] * h[lane + 32] + shift[lane + 32];
  float m = fmaxf(x0, x1);
#pragma unroll
  for (int o = 16; o > 0; o >>= 1) m = fmaxf(m, __shfl_xor(m, o, 32));
  float e0 = __expf(x0 - m), e1 = __expf(x1 - m);
  float s = e0 + e1;
#pragma unroll
  for (int o = 16; o > 0; o >>= 1) s += __shfl_xor(s, o, 32);
  float inv = 1.0f / s;
  out[(size_t)row * 64 + lane]      = e0 * inv;
  out[(size_t)row * 64 + lane + 32] = e1 * inv;
}

// ---------------------------------------------------------------------------
// Host-side launch
// ---------------------------------------------------------------------------
extern "C" void kernel_launch(void* const* d_in, const int* in_sizes, int n_in,
                              void* d_out, int out_size, void* d_ws, size_t ws_size,
                              hipStream_t stream) {
  const float* x   = (const float*)d_in[0];
  const int*   ei  = (const int*)  d_in[1];
  const float* W1a = (const float*)d_in[9];
  const float* b1a = (const float*)d_in[10];
  const float* W1b = (const float*)d_in[11];
  const float* b1b = (const float*)d_in[12];
  const float* g1  = (const float*)d_in[13];
  const float* be1 = (const float*)d_in[14];
  const float* W2a = (const float*)d_in[15];
  const float* b2a = (const float*)d_in[16];
  const float* W2b = (const float*)d_in[17];
  const float* b2b = (const float*)d_in[18];
  const float* g2  = (const float*)d_in[19];
  const float* be2 = (const float*)d_in[20];

  const int N = in_sizes[0] / IN_DIM;
  const int E = in_sizes[1] / 2;
  const int* src = ei;
  const int* dst = ei + E;

  char* base = (char*)d_ws;
  size_t off = 0;
  auto alloc = [&](size_t bytes) {
    size_t r = off; off += (bytes + 255) & ~(size_t)255; return r;
  };
  // packed weights (bf16)
  size_t oW1a = alloc((size_t)IN_DIM * HIDDIM * 2);
  size_t oW1b = alloc((size_t)HIDDIM * HIDDIM * 2);
  size_t oW2a = alloc((size_t)HIDDIM * HIDDIM * 2);
  size_t oW2b = alloc((size_t)HIDDIM * NCLS * 2);
  // BN scratch
  size_t oSum1 = alloc(HIDDIM * 4), oSq1 = alloc(HIDDIM * 4);
  size_t oSc1  = alloc(HIDDIM * 4), oSh1 = alloc(HIDDIM * 4);
  size_t oSum2 = alloc(NCLS * 4),   oSq2 = alloc(NCLS * 4);
  size_t oSc2  = alloc(NCLS * 4),   oSh2 = alloc(NCLS * 4);
  // activations
  size_t oA1  = alloc((size_t)N * IN_DIM * 4);   // f32 agg1; later reused for H2
  size_t oZ   = alloc((size_t)N * HIDDIM * 2);   // bf16 hidden (both layers)
  size_t oH1  = alloc((size_t)N * HIDDIM * 2);   // bf16 selu(layer1)
  size_t oA2  = alloc((size_t)N * HIDDIM * 4);   // f32 agg2
  (void)ws_size;

  uint32_t* pW1a = (uint32_t*)(base + oW1a);
  uint32_t* pW1b = (uint32_t*)(base + oW1b);
  uint32_t* pW2a = (uint32_t*)(base + oW2a);
  uint32_t* pW2b = (uint32_t*)(base + oW2b);
  float*    sum1 = (float*)(base + oSum1); float* sq1 = (float*)(base + oSq1);
  float*    sc1  = (float*)(base + oSc1);  float* sh1 = (float*)(base + oSh1);
  float*    sum2 = (float*)(base + oSum2); float* sq2 = (float*)(base + oSq2);
  float*    sc2  = (float*)(base + oSc2);  float* sh2 = (float*)(base + oSh2);
  float*    A1   = (float*)(base + oA1);
  uint16_t* Z    = (uint16_t*)(base + oZ);
  uint16_t* H1   = (uint16_t*)(base + oH1);
  float*    A2   = (float*)(base + oA2);
  float*    H2   = (float*)(base + oA1);          // reuse A1 region (dead by then)
  float*    out  = (float*)d_out;

  const int tilesM  = (N + 15) / 16;
  const dim3 blk(256);

  // 0) zero BN accumulators (sum1/sq1 contiguous, sum2/sq2 contiguous)
  zero_f32<<<(2 * HIDDIM + 255) / 256, blk, 0, stream>>>(sum1, 2 * HIDDIM);
  zero_f32<<<1, 128, 0, stream>>>(sum2, 2 * NCLS);

  // 1) pack weights to WMMA fragment order
  pack_w_kernel<<<(IN_DIM * HIDDIM / 2) / 256, blk, 0, stream>>>(W1a, pW1a, IN_DIM, HIDDIM);
  pack_w_kernel<<<(HIDDIM * HIDDIM / 2) / 256, blk, 0, stream>>>(W1b, pW1b, HIDDIM, HIDDIM);
  pack_w_kernel<<<(HIDDIM * HIDDIM / 2) / 256, blk, 0, stream>>>(W2a, pW2a, HIDDIM, HIDDIM);
  pack_w_kernel<<<(HIDDIM * NCLS / 2) / 256, blk, 0, stream>>>(W2b, pW2b, HIDDIM, NCLS);

  // 2) layer-1 aggregation: A1 = x ; A1[dst] += x[src]
  {
    long n4 = (long)N * IN_DIM / 4;
    copy_f32x4<<<(unsigned)((n4 + 255) / 256), blk, 0, stream>>>(
        (const float4*)x, (float4*)A1, n4);
    long t = (long)E * (IN_DIM / 4);
    scatter_f32<<<(unsigned)((t + 255) / 256), blk, 0, stream>>>(x, src, dst, A1, E, IN_DIM);
  }

  // 3) Z = relu(A1 @ W1a + b1a)   [f32 in, bf16 out]
  gemm_bias_act<IN_DIM, HIDDIM, 0, 0, 1><<<dim3(HIDDIM / 16, (tilesM + 7) / 8), blk, 0, stream>>>(
      A1, pW1a, b1a, Z, N);
  // 4) H1 = selu(Z @ W1b + b1b)   [bf16 in, bf16 out]
  gemm_bias_act<HIDDIM, HIDDIM, 1, 1, 1><<<dim3(HIDDIM / 16, (tilesM + 7) / 8), blk, 0, stream>>>(
      Z, pW1b, b1b, H1, N);

  // 5) BN1 stats + scale/shift
  col_stats_bf16<<<(HIDDIM * 128) / 256, blk, 0, stream>>>(H1, sum1, sq1, N, HIDDIM, 128);
  bn_finalize<<<(HIDDIM + 255) / 256, blk, 0, stream>>>(sum1, sq1, g1, be1, sc1, sh1,
                                                        HIDDIM, 1.0f / (float)N);

  // 6) layer-2 aggregation on normalized features: A2 = bn(H1); A2[dst] += bn(H1[src])
  {
    long total = (long)N * HIDDIM;
    bn_apply_init<<<(unsigned)((total / 4 + 255) / 256), blk, 0, stream>>>(
        H1, sc1, sh1, A2, total, HIDDIM);
    long t = (long)E * (HIDDIM / 4);
    scatter_bn_bf16<<<(unsigned)((t + 255) / 256), blk, 0, stream>>>(
        H1, src, dst, sc1, sh1, A2, E, HIDDIM);
  }

  // 7) Z = relu(A2 @ W2a + b2a) ; H2 = selu(Z @ W2b + b2b)
  gemm_bias_act<HIDDIM, HIDDIM, 0, 0, 1><<<dim3(HIDDIM / 16, (tilesM + 7) / 8), blk, 0, stream>>>(
      A2, pW2a, b2a, Z, N);
  gemm_bias_act<HIDDIM, NCLS, 1, 1, 0><<<dim3(NCLS / 16, (tilesM + 7) / 8), blk, 0, stream>>>(
      Z, pW2b, b2b, H2, N);

  // 8) BN2 stats + fused BN+softmax to output
  col_stats_f32<<<(NCLS * 256) / 256, blk, 0, stream>>>(H2, sum2, sq2, N, NCLS, 256);
  bn_finalize<<<1, 64, 0, stream>>>(sum2, sq2, g2, be2, sc2, sh2, NCLS, 1.0f / (float)N);
  bn_softmax64<<<(N + 7) / 8, blk, 0, stream>>>(H2, sc2, sh2, out, N);
}
